// MLP_GRU_63221918597465
// MI455X (gfx1250) — compile-verified, matrix operand order checked by reading, not run on hardware
//
#include <hip/hip_runtime.h>
#include <hip/hip_bf16.h>

typedef __attribute__((ext_vector_type(16))) _Float16 v16h;
typedef __attribute__((ext_vector_type(8)))  float    v8f;

#define B_  16384
#define T_  288
#define H_  512
#define LDA 520          // padded LDS row stride (halfs): 1040B = 16B mod 256 -> no bank conflicts

union Frag { v16h v; uint4 u[2]; };

__device__ __forceinline__ float fast_sigmoid(float x) {
    return 1.f / (1.f + __expf(-x));
}
__device__ __forceinline__ float fast_tanh(float x) {
    x = fminf(fmaxf(x, -15.f), 15.f);
    float e = __expf(-2.f * x);
    return (1.f - e) / (1.f + e);
}

// ---------------- prep: weight convert / transpose / h0 zero ----------------

__global__ void __launch_bounds__(256) k_prep_w1t(const float* __restrict__ w1,
                                                  _Float16* __restrict__ w1T) {
    int id = blockIdx.x * 256 + threadIdx.x;      // 512*512
    int n = id >> 9, k = id & 511;
    w1T[id] = (_Float16)w1[k * 512 + n];          // (K,N) -> (N,K)
}

__global__ void __launch_bounds__(256) k_prep_wh(const float* __restrict__ whh,
                                                 _Float16* __restrict__ whT) {
    int id = blockIdx.x * 256 + threadIdx.x;      // 1536*512, already (N,K)
    whT[id] = (_Float16)whh[id];
}

__global__ void __launch_bounds__(256) k_zero(uint4* __restrict__ p) {
    int id = blockIdx.x * 256 + threadIdx.x;
    p[id] = make_uint4(0u, 0u, 0u, 0u);
}

// ---------------- phase 1: MLP (layer1 + WMMA layer2 + gamma + x_im) --------
// Block = 128 rows (8 M-tiles), 8 waves; waves split N (4 N-tiles each).
// Each B-fragment load feeds 8 WMMAs (one per M-tile).
// K-loop is kept ROLLED (unroll 2) so A-fragment LDS loads cannot be
// hoisted/spilled across the nt loop (round-2 scratch-spill pathology).

#define MR_MLP 128
#define MT_MLP 8

__global__ void __launch_bounds__(256) k_mlp(
    const float* __restrict__ inp,
    const float* __restrict__ w0, const float* __restrict__ b0,
    const float* __restrict__ b1,
    const float* __restrict__ gxw, const float* __restrict__ gxb,
    const _Float16* __restrict__ w1T,
    float* __restrict__ x_im)
{
    __shared__ float    featX[MR_MLP][8];       // 4 KB
    __shared__ _Float16 h1s[MR_MLP * LDA];      // 130 KB (padded rows)
    __shared__ float    red16[MR_MLP][128];     // 64 KB (deterministic gamma reduce)

    const int  tid  = threadIdx.x;
    const int  lane = tid & 31;
    const int  wave = tid >> 5;
    const int  l15  = lane & 15;
    const int  hsel = lane >> 4;
    const long rowBase = (long)blockIdx.x * MR_MLP;

    if (tid < MR_MLP) {
        long row = rowBase + tid;
        int  b = (int)(row / T_);
        int  t = (int)(row % T_);
        const float* base = inp + (size_t)b * 6 * T_ + t;
        float xs = base[0 * T_];
        if (xs != xs) xs = -1.0f;               // NaN -> -1
        featX[tid][0] = (float)t;
        featX[tid][1] = base[3 * T_];           // d_fwd
        featX[tid][2] = base[5 * T_];           // d_bwd
        featX[tid][3] = xs;
        featX[tid][4] = base[1 * T_];           // x_last
        featX[tid][5] = base[2 * T_];           // mask
        featX[tid][6] = base[4 * T_];           // x_next
    }
    __syncthreads();

    // layer 1: (128,3)@(3,512) in VALU -> LDS f16 (A operand, padded rows)
    for (int i = tid; i < MR_MLP * 512; i += 256) {
        int m = i >> 9, j = i & 511;
        float v = featX[m][0] * w0[j] + featX[m][1] * w0[512 + j]
                + featX[m][2] * w0[1024 + j] + b0[j];
        h1s[m * LDA + j] = (_Float16)fmaxf(v, 0.f);
    }
    __syncthreads();

    // layer 2 + fused gamma partial dot
    float part[MT_MLP][8];
    #pragma unroll
    for (int mt = 0; mt < MT_MLP; ++mt)
        #pragma unroll
        for (int v = 0; v < 8; ++v) part[mt][v] = 0.f;

    const v8f z8 = {};
    #pragma unroll 1
    for (int nt = 0; nt < 4; ++nt) {
        int n = (wave * 4 + nt) * 16 + l15;
        const _Float16* bBase = w1T + (size_t)n * 512 + hsel * 8;
        v8f acc[MT_MLP];
        #pragma unroll
        for (int mt = 0; mt < MT_MLP; ++mt) acc[mt] = z8;

        #pragma unroll 2
        for (int ks = 0; ks < 16; ++ks) {
            Frag bf;
            bf.u[0] = *(const uint4*)(bBase + ks * 32);
            bf.u[1] = *(const uint4*)(bBase + ks * 32 + 16);
            #pragma unroll
            for (int mt = 0; mt < MT_MLP; ++mt) {
                const _Float16* aB = h1s + (mt * 16 + l15) * LDA + hsel * 8 + ks * 32;
                Frag a;
                a.u[0] = *(const uint4*)(aB);
                a.u[1] = *(const uint4*)(aB + 16);
                acc[mt] = __builtin_amdgcn_wmma_f32_16x16x32_f16(
                    false, a.v, false, bf.v, (short)0, acc[mt], false, false);
            }
        }
        float bias = b1[n];
        float gw   = gxw[n];
        #pragma unroll
        for (int mt = 0; mt < MT_MLP; ++mt)
            #pragma unroll
            for (int v = 0; v < 8; ++v)
                part[mt][v] += fmaxf(acc[mt][v] + bias, 0.f) * gw;
    }

    // deposit per-(row, lane-column) partials, fixed order reduce
    #pragma unroll
    for (int mt = 0; mt < MT_MLP; ++mt)
        #pragma unroll
        for (int v = 0; v < 8; ++v)
            red16[mt * 16 + v + 8 * hsel][wave * 16 + l15] = part[mt][v];
    __syncthreads();

    if (tid < MR_MLP) {
        float g = 0.f;
        for (int i = 0; i < 128; ++i) g += red16[tid][i];
        float gamma = fast_sigmoid(g + gxb[0]);
        float xs = featX[tid][3], xl = featX[tid][4];
        float mk = featX[tid][5], xn = featX[tid][6];
        x_im[rowBase + tid] = mk * xs + (1.f - mk) * (gamma * xl + (1.f - gamma) * xn);
    }
}

// ---------------- phase 2: one GRU step ----------------
// Grid (B/128, 4). Block = 128 rows (8 M-tiles) x 8 j-triples (one per wave).
// Each B-fragment (3 gates) feeds 24 WMMAs; A staged in LDS; h_old read from LDS.

#define MR_GRU 128
#define MT_GRU 8

__global__ void __launch_bounds__(256) k_gru_step(
    const _Float16* __restrict__ h_in,
    _Float16* __restrict__ h_out,
    const float* __restrict__ x_im,
    const float* __restrict__ inp,
    const float* __restrict__ Wih, const float* __restrict__ bih,
    const float* __restrict__ bhh,
    const _Float16* __restrict__ whT,
    int t)
{
    __shared__ _Float16 hsm[MR_GRU * LDA];      // 130 KB
    __shared__ float xim_s[MR_GRU], mk_s[MR_GRU];

    const int tid   = threadIdx.x;
    const int lane  = tid & 31;
    const int wave  = tid >> 5;
    const int l15   = lane & 15;
    const int hsel  = lane >> 4;
    const int mBase = blockIdx.x * MR_GRU;

    if (tid < MR_GRU) {
        int b = mBase + tid;
        xim_s[tid] = x_im[(size_t)b * T_ + t];
        mk_s[tid]  = inp[(size_t)b * 6 * T_ + 2 * T_ + t];
    }
    {   // stage A-tile (128 x 512 f16) into padded LDS rows
        const uint4* src = (const uint4*)(h_in + (size_t)mBase * 512);
        for (int i = tid; i < MR_GRU * 64; i += 256) {
            int r = i >> 6, c = i & 63;
            *(uint4*)(hsm + r * LDA + c * 8) = src[r * 64 + c];
        }
    }
    __syncthreads();

    const int j = (blockIdx.y * 8 + wave) * 16 + l15;   // this wave's gate column
    const _Float16* bR = whT + (size_t)j * 512 + hsel * 8;
    const _Float16* bZ = bR + (size_t)512 * 512;
    const _Float16* bN = bZ + (size_t)512 * 512;

    v8f accR[MT_GRU], accZ[MT_GRU], accN[MT_GRU];
    const v8f z8 = {};
    #pragma unroll
    for (int mt = 0; mt < MT_GRU; ++mt) { accR[mt] = z8; accZ[mt] = z8; accN[mt] = z8; }

    #pragma unroll 2
    for (int ks = 0; ks < 16; ++ks) {
        Frag fr, fz, fn;
        fr.u[0] = *(const uint4*)(bR + ks * 32); fr.u[1] = *(const uint4*)(bR + ks * 32 + 16);
        fz.u[0] = *(const uint4*)(bZ + ks * 32); fz.u[1] = *(const uint4*)(bZ + ks * 32 + 16);
        fn.u[0] = *(const uint4*)(bN + ks * 32); fn.u[1] = *(const uint4*)(bN + ks * 32 + 16);
        #pragma unroll
        for (int mt = 0; mt < MT_GRU; ++mt) {
            const _Float16* aB = hsm + (mt * 16 + l15) * LDA + hsel * 8 + ks * 32;
            Frag a;
            a.u[0] = *(const uint4*)(aB);
            a.u[1] = *(const uint4*)(aB + 16);
            accR[mt] = __builtin_amdgcn_wmma_f32_16x16x32_f16(
                false, a.v, false, fr.v, (short)0, accR[mt], false, false);
            accZ[mt] = __builtin_amdgcn_wmma_f32_16x16x32_f16(
                false, a.v, false, fz.v, (short)0, accZ[mt], false, false);
            accN[mt] = __builtin_amdgcn_wmma_f32_16x16x32_f16(
                false, a.v, false, fn.v, (short)0, accN[mt], false, false);
        }
    }

    // gate params depend only on j (W_ih folded in; gi never materialized)
    float w0r = Wih[2 * j],          w1r = Wih[2 * j + 1];
    float w0z = Wih[2 * (j + 512)],  w1z = Wih[2 * (j + 512) + 1];
    float w0n = Wih[2 * (j + 1024)], w1n = Wih[2 * (j + 1024) + 1];
    float brz = bih[j] + bhh[j];
    float bzz = bih[j + 512] + bhh[j + 512];
    float bin = bih[j + 1024];
    float bhn = bhh[j + 1024];

    #pragma unroll
    for (int mt = 0; mt < MT_GRU; ++mt) {
        #pragma unroll
        for (int v = 0; v < 8; ++v) {
            int mloc = mt * 16 + v + 8 * hsel;
            float x = xim_s[mloc], mk = mk_s[mloc];
            float r  = fast_sigmoid(accR[mt][v] + w0r * x + w1r * mk + brz);
            float z  = fast_sigmoid(accZ[mt][v] + w0z * x + w1z * mk + bzz);
            float nn = fast_tanh(w0n * x + w1n * mk + bin + r * (accN[mt][v] + bhn));
            float hold = (float)hsm[mloc * LDA + j];
            h_out[(size_t)(mBase + mloc) * 512 + j] = (_Float16)((1.f - z) * nn + z * hold);
        }
    }
}

// ---------------- final FC ----------------

__global__ void __launch_bounds__(256) k_fc(const _Float16* __restrict__ h,
                                            const float* __restrict__ fcw,
                                            const float* __restrict__ fcb,
                                            float* __restrict__ out) {
    int b = blockIdx.x * 256 + threadIdx.x;
    const _Float16* row = h + (size_t)b * 512;
    float s = 0.f;
    #pragma unroll 8
    for (int jj = 0; jj < 512; ++jj) s += (float)row[jj] * fcw[jj];
    out[b] = s + fcb[0];
}

// ---------------- host ----------------

extern "C" void kernel_launch(void* const* d_in, const int* in_sizes, int n_in,
                              void* d_out, int out_size, void* d_ws, size_t ws_size,
                              hipStream_t stream) {
    const float* inp = (const float*)d_in[0];
    const float* w0  = (const float*)d_in[1];
    const float* b0  = (const float*)d_in[2];
    const float* w1  = (const float*)d_in[3];
    const float* b1  = (const float*)d_in[4];
    const float* gxw = (const float*)d_in[5];
    const float* gxb = (const float*)d_in[6];
    const float* Wih = (const float*)d_in[7];
    const float* bih = (const float*)d_in[8];
    const float* Whh = (const float*)d_in[9];
    const float* bhh = (const float*)d_in[10];
    const float* fcw = (const float*)d_in[11];
    const float* fcb = (const float*)d_in[12];
    float* out = (float*)d_out;

    char* ws = (char*)d_ws;
    size_t off = 0;
    auto alloc = [&](size_t bytes) {
        char* p = ws + off;
        off += (bytes + 255) & ~size_t(255);
        return p;
    };
    _Float16* w1T = (_Float16*)alloc((size_t)512 * 512 * 2);      // 0.5 MB
    _Float16* whT = (_Float16*)alloc((size_t)1536 * 512 * 2);     // 1.5 MB
    float*    xim = (float*)   alloc((size_t)B_ * T_ * 4);        // 18.9 MB
    _Float16* hA  = (_Float16*)alloc((size_t)B_ * 512 * 2);       // 16 MB
    _Float16* hB  = (_Float16*)alloc((size_t)B_ * 512 * 2);       // 16 MB
    (void)ws_size; (void)in_sizes; (void)n_in; (void)out_size;

    k_prep_w1t<<<512 * 512 / 256, 256, 0, stream>>>(w1, w1T);
    k_prep_wh<<<1536 * 512 / 256, 256, 0, stream>>>(Whh, whT);
    k_zero<<<(B_ * 512 / 8) / 256, 256, 0, stream>>>((uint4*)hA);

    k_mlp<<<(unsigned)((long)B_ * T_ / MR_MLP), 256, 0, stream>>>(
        inp, w0, b0, b1, gxw, gxb, w1T, xim);

    for (int t = 0; t < T_; ++t) {
        const _Float16* hin = (t & 1) ? hB : hA;
        _Float16* hout      = (t & 1) ? hA : hB;
        k_gru_step<<<dim3(B_ / MR_GRU, 4), 256, 0, stream>>>(
            hin, hout, xim, inp, Wih, bih, bhh, whT, t);
    }
    // after t=287 (odd), final h lives in hA
    k_fc<<<B_ / 256, 256, 0, stream>>>(hA, fcw, fcb, out);
}
